// SPModel_40278203301978
// MI455X (gfx1250) — matrix operand order repaired and checked by reading, not compiled, not run on hardware
//
#include <hip/hip_runtime.h>
#include <hip/hip_bf16.h>

// ---------------------------------------------------------------------------
// Model dims (fixed by the reference)
// ---------------------------------------------------------------------------
#define BB   4
#define TT   1024
#define DD   1024
#define HH   16
#define LL   12
#define RR   32
#define HDIM 64
#define MTOT (BB * TT)          // 4096 token rows

typedef __attribute__((ext_vector_type(16))) __bf16 v16bf;
typedef __attribute__((ext_vector_type(4)))  __bf16 v4bf;
typedef __attribute__((ext_vector_type(8)))  float  v8f;

// ---------------------------------------------------------------------------
// Embedding: h[b,t,:] = wte[ids[b,t],:] + wpe[t,:]
// ---------------------------------------------------------------------------
__global__ __launch_bounds__(256) void embed_kernel(
    const int* __restrict__ ids, const float* __restrict__ wte,
    const float* __restrict__ wpe, float* __restrict__ h)
{
    int idx = blockIdx.x * 256 + threadIdx.x;     // over B*T*D
    if (idx >= MTOT * DD) return;
    int d  = idx % DD;
    int bt = idx / DD;
    int t  = bt % TT;
    h[idx] = wte[(size_t)ids[bt] * DD + d] + wpe[(size_t)t * DD + d];
}

// ---------------------------------------------------------------------------
// LayerNorm over D=1024: one 256-thread block per row, 4 elems/thread
// ---------------------------------------------------------------------------
__global__ __launch_bounds__(256) void ln_kernel(
    const float* __restrict__ x, const float* __restrict__ g,
    const float* __restrict__ b, float* __restrict__ y)
{
    __shared__ float red[256];
    int row = blockIdx.x;
    const float* xr = x + (size_t)row * DD;

    float vals[4];
    float s = 0.f;
    for (int i = 0; i < 4; i++) {
        vals[i] = xr[threadIdx.x + i * 256];
        s += vals[i];
    }
    red[threadIdx.x] = s;
    __syncthreads();
    for (int off = 128; off > 0; off >>= 1) {
        if (threadIdx.x < off) red[threadIdx.x] += red[threadIdx.x + off];
        __syncthreads();
    }
    float mean = red[0] * (1.0f / DD);
    __syncthreads();

    float v = 0.f;
    for (int i = 0; i < 4; i++) {
        float d = vals[i] - mean;
        v += d * d;
    }
    red[threadIdx.x] = v;
    __syncthreads();
    for (int off = 128; off > 0; off >>= 1) {
        if (threadIdx.x < off) red[threadIdx.x] += red[threadIdx.x + off];
        __syncthreads();
    }
    float rstd = rsqrtf(red[0] * (1.0f / DD) + 1e-5f);

    for (int i = 0; i < 4; i++) {
        int d = threadIdx.x + i * 256;
        y[(size_t)row * DD + d] = (vals[i] - mean) * rstd * g[d] + b[d];
    }
}

// ---------------------------------------------------------------------------
// Exact GELU (erf form), in place
// ---------------------------------------------------------------------------
__global__ __launch_bounds__(256) void gelu_kernel(float* __restrict__ x, int n)
{
    int i = blockIdx.x * 256 + threadIdx.x;
    if (i < n) {
        float v = x[i];
        x[i] = 0.5f * v * (1.f + erff(v * 0.70710678118654752f));
    }
}

// ---------------------------------------------------------------------------
// Main WMMA GEMM:  Y[M,N] (op)= scale * (X[M,K] @ W[K,N]) [+ bias]
// Templated on <HAS_BIAS, ACCUM> so the epilogue is straight-line code
// (no per-element runtime branches; accumulate loads can batch).
// Block tile 128x128, 8 waves in a 4x2 grid; wave tile 32x64
//   = 2 A-frags x 4 B-frags = 8 v_wmma_f32_16x16x32_bf16 per k-step.
// LDS: Xs row-major [m][k]; Wt TRANSPOSED [n][k] so every fragment is 16
// contiguous bf16 (-> 2x ds_load_b128 per fragment).
// REQUIRES tile-aligned dims: M%128==0, N%128==0, K%32==0 (true at all
// call sites; N==32 LoRA GEMMs use the dedicated kernel below).
// ---------------------------------------------------------------------------
#define GBM 128
#define GBN 128
#define GBK 32

template <bool HAS_BIAS, bool ACCUM>
__global__ __launch_bounds__(256) void wmma_gemm_kernel(
    const float* __restrict__ X, const float* __restrict__ W,
    const float* __restrict__ bias, float* __restrict__ Y,
    int M, int N, int K, float scale)
{
    __shared__ __bf16 Xs[GBM * GBK];   // [m 128][k 32]
    __shared__ __bf16 Wt[GBN * GBK];   // [n 128][k 32]  (transposed W tile)

    int tid   = threadIdx.x;
    int lane  = tid & 31;
    int wave  = tid >> 5;
    int l16   = lane & 15;
    int lhalf = lane >> 4;             // 0 or 1

    int wm = (wave & 3) * 32;          // wave row offset in block tile
    int wn = (wave >> 2) * 64;         // wave col offset in block tile
    int m0 = blockIdx.y * GBM;
    int n0 = blockIdx.x * GBN;

    v8f acc[2][4] = {};

    for (int kk = 0; kk < K; kk += GBK) {
        // X tile: 1024 float4 chunks, 4 per thread (unrolled)
#pragma unroll
        for (int t = 0; t < 4; t++) {
            int i  = tid + t * 256;
            int r  = i >> 3;                 // 0..127
            int c4 = i & 7;                  // 0..7 (groups of 4 k)
            const float* xp = X + (size_t)(m0 + r) * K + kk + c4 * 4;
            const float4 xv = *reinterpret_cast<const float4*>(xp);
            v4bf p;
            p[0] = (__bf16)xv.x; p[1] = (__bf16)xv.y;
            p[2] = (__bf16)xv.z; p[3] = (__bf16)xv.w;
            *reinterpret_cast<v4bf*>(&Xs[r * GBK + c4 * 4]) = p;
            __builtin_prefetch(xp + GBK, 0, 1);       // speculative, next k-tile
        }
        // W tile, stored transposed: thread handles one column, 4 k-vals
        // (lanes read consecutive n for fixed k -> coalesced)
#pragma unroll
        for (int t = 0; t < 4; t++) {
            int i   = tid + t * 256;
            int col = i & 127;               // n within tile
            int kg  = i >> 7;                // 0..7 (groups of 4 k)
            const float* wp = W + (size_t)(kk + kg * 4) * N + n0 + col;
            v4bf p;
            p[0] = (__bf16)wp[0];
            p[1] = (__bf16)wp[(size_t)N];
            p[2] = (__bf16)wp[(size_t)2 * N];
            p[3] = (__bf16)wp[(size_t)3 * N];
            *reinterpret_cast<v4bf*>(&Wt[col * GBK + kg * 4]) = p;
            __builtin_prefetch(wp + (size_t)GBK * N, 0, 1);
        }
        __syncthreads();

        // A fragments (16x32): row = l16, 16 contiguous k -> 2x b128
        v16bf a[2], b[4];
        for (int i = 0; i < 2; i++) {
            int row = wm + i * 16 + l16;
            int kb  = 16 * lhalf;
            for (int e = 0; e < 16; e++) {
                int kc = kb + e;
                a[i][e] = Xs[row * GBK + kc];
            }
        }
        // B fragments (32x16): col = l16, 16 contiguous k -> 2x b128
        for (int j = 0; j < 4; j++) {
            int col = wn + j * 16 + l16;
            for (int e = 0; e < 16; e++)
                b[j][e] = Wt[col * GBK + 16 * lhalf + e];
        }
        for (int i = 0; i < 2; i++)
            for (int j = 0; j < 4; j++)
                acc[i][j] = __builtin_amdgcn_wmma_f32_16x16x32_bf16(
                    false, a[i], false, b[j], (short)0, acc[i][j], false, false);
        __syncthreads();
    }

    // epilogue (straight-line): element p of lane -> M = p + 8*lhalf, N = l16
#pragma unroll
    for (int i = 0; i < 2; i++)
#pragma unroll
        for (int j = 0; j < 4; j++) {
            int gn = n0 + wn + j * 16 + l16;
            float bb = HAS_BIAS ? bias[gn] : 0.f;
#pragma unroll
            for (int p = 0; p < 8; p++) {
                int gm = m0 + wm + i * 16 + p + 8 * lhalf;
                float v = acc[i][j][p] * scale + bb;
                if (ACCUM) v += Y[(size_t)gm * N + gn];
                Y[(size_t)gm * N + gn] = v;
            }
        }
}

// ---------------------------------------------------------------------------
// LoRA-A WMMA GEMM (N == 32 exactly):  Y[M,32] = X[M,K] @ W[K,32]
// Block tile 256x32, 8 waves; wave tile 32x32 = 2x2 frags = 4 WMMA/k-step.
// No bounds checks anywhere (M%256==0, K%32==0 at all call sites).
// ---------------------------------------------------------------------------
#define LBM 256

__global__ __launch_bounds__(256) void wmma_gemm_n32_kernel(
    const float* __restrict__ X, const float* __restrict__ W,
    float* __restrict__ Y, int M, int K)
{
    __shared__ __bf16 Xs[LBM * GBK];   // [m 256][k 32]
    __shared__ __bf16 Wt[RR * GBK];    // [n 32][k 32]  (transposed)

    int tid   = threadIdx.x;
    int lane  = tid & 31;
    int wave  = tid >> 5;
    int l16   = lane & 15;
    int lhalf = lane >> 4;

    int wm = wave * 32;                // wave row offset in block tile
    int m0 = blockIdx.x * LBM;

    v8f acc[2][2] = {};

    for (int kk = 0; kk < K; kk += GBK) {
        // X tile: 2048 float4 chunks, 8 per thread (unrolled)
#pragma unroll
        for (int t = 0; t < 8; t++) {
            int i  = tid + t * 256;
            int r  = i >> 3;                 // 0..255
            int c4 = i & 7;
            const float* xp = X + (size_t)(m0 + r) * K + kk + c4 * 4;
            const float4 xv = *reinterpret_cast<const float4*>(xp);
            v4bf p;
            p[0] = (__bf16)xv.x; p[1] = (__bf16)xv.y;
            p[2] = (__bf16)xv.z; p[3] = (__bf16)xv.w;
            *reinterpret_cast<v4bf*>(&Xs[r * GBK + c4 * 4]) = p;
            __builtin_prefetch(xp + GBK, 0, 1);
        }
        // W tile 32x32, transposed store; 256 float4 chunks; thread handles 1
        {
            int col = tid & 31;              // n
            int kg  = tid >> 5;              // 0..7 (groups of 4 k)
            const float* wp = W + (size_t)(kk + kg * 4) * RR + col;
            v4bf p;
            p[0] = (__bf16)wp[0];
            p[1] = (__bf16)wp[RR];
            p[2] = (__bf16)wp[2 * RR];
            p[3] = (__bf16)wp[3 * RR];
            *reinterpret_cast<v4bf*>(&Wt[col * GBK + kg * 4]) = p;
        }
        __syncthreads();

        v16bf a[2], b[2];
        for (int i = 0; i < 2; i++) {
            int row = wm + i * 16 + l16;
            int kb  = 16 * lhalf;
            for (int e = 0; e < 16; e++) {
                int kc = kb + e;
                a[i][e] = Xs[row * GBK + kc];
            }
        }
        for (int j = 0; j < 2; j++) {
            int col = j * 16 + l16;
            for (int e = 0; e < 16; e++)
                b[j][e] = Wt[col * GBK + 16 * lhalf + e];
        }
        for (int i = 0; i < 2; i++)
            for (int j = 0; j < 2; j++)
                acc[i][j] = __builtin_amdgcn_wmma_f32_16x16x32_bf16(
                    false, a[i], false, b[j], (short)0, acc[i][j], false, false);
        __syncthreads();
    }

#pragma unroll
    for (int i = 0; i < 2; i++)
#pragma unroll
        for (int j = 0; j < 2; j++)
#pragma unroll
            for (int p = 0; p < 8; p++) {
                int gm = m0 + wm + i * 16 + p + 8 * lhalf;
                int gn = j * 16 + l16;
                Y[(size_t)gm * RR + gn] = acc[i][j][p];
            }
}

// ---------------------------------------------------------------------------
// Fused causal attention with inline KV fake-quant (8-bit, asymmetric).
// qkv: [B,T,3D]; out: [B,T,D]. Block = 128 thr (4 waves); wave w handles
// q rows q0 = bx*64 + w*16. Flash-style over 32-key chunks:
//   S = Q@K^T (WMMA), online softmax in per-wave LDS, O += P@V (WMMA).
// Ks row-major [key][hd] (S B-frags contiguous); Vs TRANSPOSED [hd][key]
// (P@V B-frags contiguous) -> all fragments lower to ds_load_b128.
// K/V staged with float4 loads; fake-quant uses a hoisted reciprocal.
// ---------------------------------------------------------------------------
__global__ __launch_bounds__(128) void attn_kernel(
    const float* __restrict__ qkv, float* __restrict__ out,
    const float* __restrict__ kvmin, const float* __restrict__ kvmax)
{
    __shared__ __bf16 Ks[32 * HDIM];        // [key 32][hd 64]
    __shared__ __bf16 Vs[HDIM * 32];        // [hd 64][key 32]  (transposed)
    __shared__ float  Sld[4][16 * 32];      // per-wave score tile
    __shared__ __bf16 Pld[4][16 * 32];      // per-wave probs (bf16)
    __shared__ float  Mrow[4][16], Lrow[4][16], Arow[4][16];

    int tid   = threadIdx.x;
    int lane  = tid & 31;
    int wave  = tid >> 5;
    int l16   = lane & 15;
    int lhalf = lane >> 4;

    int bh = blockIdx.y;                    // b*H + h
    int b  = bh / HH, hh = bh % HH;
    int q0 = blockIdx.x * 64 + wave * 16;

    float mn = kvmin[0], mx = kvmax[0];
    float qs  = (mx - mn) * (1.0f / 255.0f);
    float rqs = 255.0f / (mx - mn);         // hoisted reciprocal of qs
    float zp  = rintf(-mn * rqs);
    const float sm_scale = 0.125f;          // 1/sqrt(64)

    // Q fragments (A 16x32 bf16): rows q0..q0+15, hd 0..63 -> 2 frags
    v16bf qf[2];
    {
        int row = q0 + l16;
        const float* qp = qkv + ((size_t)(b * TT + row)) * (3 * DD) + hh * HDIM;
        int kb = 16 * lhalf;
        for (int f = 0; f < 2; f++)
            for (int e = 0; e < 16; e++) {
                int hd = 32 * f + kb + e;
                qf[f][e] = (__bf16)qp[hd];
            }
    }

    v8f oacc[4] = {};                       // hd tiles 0,16,32,48
    if (lhalf == 0) { Mrow[wave][l16] = -1e30f; Lrow[wave][l16] = 0.f; }

    int nchunk = (blockIdx.x + 1) * 2;      // keys 0 .. bx*64+63, chunks of 32
    for (int kc = 0; kc < nchunk; kc++) {
        int kb_g = kc * 32;

        // stage K/V chunk with fake-quant: 512 float4 chunks, 4 per thread
#pragma unroll
        for (int t = 0; t < 4; t++) {
            int i  = tid + t * 128;          // 0..511
            int kl = i >> 4;                 // 0..31 key
            int h4 = (i & 15) * 4;           // 0..60 hd group of 4
            size_t base = ((size_t)(b * TT + kb_g + kl)) * (3 * DD) + hh * HDIM + h4;
            const float4 kv4 = *reinterpret_cast<const float4*>(qkv + base + DD);
            const float4 vv4 = *reinterpret_cast<const float4*>(qkv + base + 2 * DD);
            float kf[4] = {kv4.x, kv4.y, kv4.z, kv4.w};
            float vf[4] = {vv4.x, vv4.y, vv4.z, vv4.w};
            v4bf kp;
#pragma unroll
            for (int q = 0; q < 4; q++) {
                float kq = fminf(fmaxf(rintf(kf[q] * rqs) + zp, 0.f), 255.f);
                float vq = fminf(fmaxf(rintf(vf[q] * rqs) + zp, 0.f), 255.f);
                kp[q] = (__bf16)((kq - zp) * qs);
                Vs[(h4 + q) * 32 + kl] = (__bf16)((vq - zp) * qs);
            }
            *reinterpret_cast<v4bf*>(&Ks[kl * HDIM + h4]) = kp;
        }
        __syncthreads();

        // S(16x32) = Q(16x64) @ K^T(64x32): 2 key-tiles x 2 hd-frags
        v8f sacc[2] = {};
        for (int j = 0; j < 2; j++) {
            for (int f = 0; f < 2; f++) {
                v16bf bf;
                int key = j * 16 + l16;
                for (int e = 0; e < 16; e++)
                    bf[e] = Ks[key * HDIM + 32 * f + 16 * lhalf + e];
                sacc[j] = __builtin_amdgcn_wmma_f32_16x16x32_bf16(
                    false, qf[f], false, bf, (short)0, sacc[j], false, false);
            }
        }
        // dump S to per-wave LDS
        for (int j = 0; j < 2; j++)
            for (int p = 0; p < 8; p++)
                Sld[wave][(p + 8 * lhalf) * 32 + j * 16 + l16] = sacc[j][p];
        __syncthreads();

        // online softmax: lanes 0..15 each own one row
        if (lhalf == 0) {
            int row  = l16;
            int qidx = q0 + row;
            float mold = Mrow[wave][row];
            float mnew = mold;
            for (int c = 0; c < 32; c++) {
                int kidx = kb_g + c;
                float v = (kidx <= qidx) ? Sld[wave][row * 32 + c] * sm_scale : -1e30f;
                mnew = fmaxf(mnew, v);
            }
            float alpha = __expf(mold - mnew);
            float ls = Lrow[wave][row] * alpha;
            for (int c = 0; c < 32; c++) {
                int kidx = kb_g + c;
                float v = (kidx <= qidx) ? Sld[wave][row * 32 + c] * sm_scale : -1e30f;
                float p = __expf(v - mnew);
                Pld[wave][row * 32 + c] = (__bf16)p;
                ls += p;
            }
            Mrow[wave][row] = mnew;
            Lrow[wave][row] = ls;
            Arow[wave][row] = alpha;
        }
        __syncthreads();

        // rescale O accumulators by alpha[row]
        for (int t = 0; t < 4; t++)
            for (int p = 0; p < 8; p++)
                oacc[t][p] *= Arow[wave][p + 8 * lhalf];

        // P fragment (A 16x32): 16 contiguous -> 2x b128
        v16bf pf;
        {
            int row = l16;
            int kb  = 16 * lhalf;
            for (int e = 0; e < 16; e++) {
                int c = kb + e;
                pf[e] = Pld[wave][row * 32 + c];
            }
        }
        // O(16x64) += P(16x32) @ V(32x64); V-frags contiguous in Vs^T
        for (int t = 0; t < 4; t++) {
            v16bf bv;
            int hd = t * 16 + l16;
            for (int e = 0; e < 16; e++)
                bv[e] = Vs[hd * 32 + 16 * lhalf + e];
            oacc[t] = __builtin_amdgcn_wmma_f32_16x16x32_bf16(
                false, pf, false, bv, (short)0, oacc[t], false, false);
        }
        __syncthreads();   // protect Ks/Vs/Sld/Pld before next chunk
    }

    // final: divide by softmax denominator, write out[b, q, h*HD + hd]
    for (int t = 0; t < 4; t++)
        for (int p = 0; p < 8; p++) {
            int r = p + 8 * lhalf;
            float denom = Lrow[wave][r];
            out[((size_t)(b * TT + q0 + r)) * DD + hh * HDIM + t * 16 + l16] =
                oacc[t][p] / denom;
        }
}

// ---------------------------------------------------------------------------
// Host-side orchestration
// ---------------------------------------------------------------------------
static inline void gemm(const float* X, const float* W, const float* bias,
                        float* Y, int M, int N, int K, float scale, int accum,
                        hipStream_t stream)
{
    dim3 grid(N / GBN, M / GBM);
    if (bias && accum)
        wmma_gemm_kernel<true, true><<<grid, 256, 0, stream>>>(X, W, bias, Y, M, N, K, scale);
    else if (bias)
        wmma_gemm_kernel<true, false><<<grid, 256, 0, stream>>>(X, W, bias, Y, M, N, K, scale);
    else if (accum)
        wmma_gemm_kernel<false, true><<<grid, 256, 0, stream>>>(X, W, bias, Y, M, N, K, scale);
    else
        wmma_gemm_kernel<false, false><<<grid, 256, 0, stream>>>(X, W, bias, Y, M, N, K, scale);
}

static inline void gemm_n32(const float* X, const float* W, float* Y,
                            int M, int K, hipStream_t stream)
{
    wmma_gemm_n32_kernel<<<M / LBM, 256, 0, stream>>>(X, W, Y, M, K);
}

extern "C" void kernel_launch(void* const* d_in, const int* in_sizes, int n_in,
                              void* d_out, int out_size, void* d_ws, size_t ws_size,
                              hipStream_t stream)
{
    const int*   ids      = (const int*)  d_in[0];
    const float* wte      = (const float*)d_in[1];
    const float* wpe      = (const float*)d_in[2];
    const float* ln1_g    = (const float*)d_in[3];
    const float* ln1_b    = (const float*)d_in[4];
    const float* attn_w   = (const float*)d_in[5];
    const float* attn_b   = (const float*)d_in[6];
    const float* attn_la  = (const float*)d_in[7];
    const float* attn_lb  = (const float*)d_in[8];
    const float* proj_w   = (const float*)d_in[9];
    const float* proj_b   = (const float*)d_in[10];
    const float* proj_la  = (const float*)d_in[11];
    const float* proj_lb  = (const float*)d_in[12];
    const float* ln2_g    = (const float*)d_in[13];
    const float* ln2_b    = (const float*)d_in[14];
    const float* fc_w     = (const float*)d_in[15];
    const float* fc_b     = (const float*)d_in[16];
    const float* fc_la    = (const float*)d_in[17];
    const float* fc_lb    = (const float*)d_in[18];
    const float* mproj_w  = (const float*)d_in[19];
    const float* mproj_b  = (const float*)d_in[20];
    const float* mproj_la = (const float*)d_in[21];
    const float* mproj_lb = (const float*)d_in[22];
    const float* kv_min   = (const float*)d_in[23];
    const float* kv_max   = (const float*)d_in[24];
    const float* lnf_g    = (const float*)d_in[25];
    const float* lnf_b    = (const float*)d_in[26];

    // workspace layout (big region shared: qkv [48MB] then mid [64MB])
    char* ws = (char*)d_ws;
    const size_t MB = 1024ull * 1024ull;
    float* h    = (float*)(ws + 0 * MB);    // 16MB  [B,T,D]
    float* xn   = (float*)(ws + 16 * MB);   // 16MB  [B,T,D]  (LN output)
    float* big  = (float*)(ws + 32 * MB);   // 64MB  qkv [B,T,3D] / mid [B,T,4D]
    float* o    = (float*)(ws + 96 * MB);   // 16MB  attention output [B,T,D]
    float* lora = (float*)(ws + 112 * MB);  // 0.5MB [B,T,R]
    (void)ws_size; (void)in_sizes; (void)n_in; (void)out_size;

    float* qkv = big;
    float* mid = big;

    embed_kernel<<<(MTOT * DD + 255) / 256, 256, 0, stream>>>(ids, wte, wpe, h);

    for (int l = 0; l < LL; l++) {
        // ---- attention sub-block ----
        ln_kernel<<<MTOT, 256, 0, stream>>>(h, ln1_g + (size_t)l * DD,
                                            ln1_b + (size_t)l * DD, xn);
        gemm_n32(xn, attn_la + (size_t)l * DD * RR, lora, MTOT, DD, stream);
        gemm(xn, attn_w + (size_t)l * DD * 3 * DD, attn_b + (size_t)l * 3 * DD,
             qkv, MTOT, 3 * DD, DD, 1.0f, 0, stream);
        gemm(lora, attn_lb + (size_t)l * RR * 3 * DD, nullptr,
             qkv, MTOT, 3 * DD, RR, 2.0f, 1, stream);

        attn_kernel<<<dim3(TT / 64, BB * HH), 128, 0, stream>>>(
            qkv, o, kv_min + l, kv_max + l);

        gemm_n32(o, proj_la + (size_t)l * DD * RR, lora, MTOT, DD, stream);
        gemm(o, proj_w + (size_t)l * DD * DD, proj_b + (size_t)l * DD,
             h, MTOT, DD, DD, 1.0f, 1, stream);                    // h += proj
        gemm(lora, proj_lb + (size_t)l * RR * DD, nullptr,
             h, MTOT, DD, RR, 2.0f, 1, stream);                    // h += lora

        // ---- MLP sub-block ----
        ln_kernel<<<MTOT, 256, 0, stream>>>(h, ln2_g + (size_t)l * DD,
                                            ln2_b + (size_t)l * DD, xn);
        gemm_n32(xn, fc_la + (size_t)l * DD * RR, lora, MTOT, DD, stream);
        gemm(xn, fc_w + (size_t)l * DD * 4 * DD, fc_b + (size_t)l * 4 * DD,
             mid, MTOT, 4 * DD, DD, 1.0f, 0, stream);
        gemm(lora, fc_lb + (size_t)l * RR * 4 * DD, nullptr,
             mid, MTOT, 4 * DD, RR, 2.0f, 1, stream);
        gelu_kernel<<<(MTOT * 4 * DD + 255) / 256, 256, 0, stream>>>(mid, MTOT * 4 * DD);

        gemm_n32(mid, mproj_la + (size_t)l * 4 * DD * RR, lora, MTOT, 4 * DD, stream);
        gemm(mid, mproj_w + (size_t)l * 4 * DD * DD, mproj_b + (size_t)l * DD,
             h, MTOT, DD, 4 * DD, 1.0f, 1, stream);                // h += mproj
        gemm(lora, mproj_lb + (size_t)l * RR * DD, nullptr,
             h, MTOT, DD, RR, 2.0f, 1, stream);                    // h += lora
    }

    ln_kernel<<<MTOT, 256, 0, stream>>>(h, lnf_g, lnf_b, (float*)d_out);
}
